// NearestNeighborTokenizer_128849018942
// MI455X (gfx1250) — matrix operand order repaired and checked by reading, not compiled, not run on hardware
//
#include <hip/hip_runtime.h>
#include <hip/hip_bf16.h>
#include <stdint.h>

typedef __attribute__((ext_vector_type(16))) __bf16 v16bf;
typedef __attribute__((ext_vector_type(8)))  float  v8f;
typedef unsigned int v4u __attribute__((ext_vector_type(4)));
typedef int          v4i __attribute__((ext_vector_type(4)));
typedef int          v8i __attribute__((ext_vector_type(8)));

#define M_TOT   4096      // B*S
#define N_TOT   16384     // MAX_CODES
#define K_DIM   256       // D
#define ROWW    (2 * K_DIM)       // interleaved row: 256 hi bf16 + 256 lo bf16
#define NSPLIT  16        // N-slices (grid.y)
#define BM      64        // queries per block = 4 waves * 16
#define TN      16        // codes per LDS tile
#define NPS     (N_TOT / NSPLIT)  // 1024 codes per slice
#define TILES   (NPS / TN)        // 64 tiles per slice
#define THRESH  512.0f

#define TILE_U16   (TN * ROWW)    // 8192 u16 elements = 16KB per tile
#define TILE_BYTES (TILE_U16 * 2)

union FragB { uint4 q[2]; v16bf v; };
static_assert(sizeof(v16bf) == 32, "v16bf must be 8 VGPRs");

// ---------------------------------------------------------------------------
// TDM: async-DMA one contiguous 16KB tile (8192 x 2-byte elements) into LDS.
// D# built per CDNA5 ISA (group0) / §8.4 (group1).
// 1D copy: tile_dim0 = tensor_dim0 = stride0 = 8192 elems, dims 1/2 unused.
// ---------------------------------------------------------------------------
__device__ __forceinline__ void tdm_load_tile(uint32_t lds_addr, uint64_t gaddr) {
  v4u g0;
  g0[0] = 1u;                                   // count=1 (valid), no gather
  g0[1] = lds_addr;                             // [63:32]  lds_addr (bytes)
  g0[2] = (uint32_t)gaddr;                      // [95:64]  global_addr[31:0]
  g0[3] = ((uint32_t)(gaddr >> 32) & 0x1FFFFFFu) | (2u << 30); // ga[56:32] | type=2
  v8i g1;
  g1[0] = 0x10000;                              // wg_mask=0, data_size=1 (2B)
  g1[1] = (int)(8192u << 16);                   // [63:48]  tensor_dim0 lo16
  g1[2] = (int)(1u << 16);                      // dim0 hi16=0 | [95:80] tensor_dim1=1
  g1[3] = (int)(8192u << 16);                   // dim1 hi16=0 | [127:112] tile_dim0
  g1[4] = 0;                                    // tile_dim1=0, tile_dim2=0 (unused)
  g1[5] = 8192;                                 // tensor_dim0_stride[31:0]
  g1[6] = 0;                                    // stride0 hi | stride1 lo
  g1[7] = 0;                                    // stride1 hi
  v4i z = {0, 0, 0, 0};
#if defined(__clang_major__) && (__clang_major__ >= 23)
  v8i z8 = {0, 0, 0, 0, 0, 0, 0, 0};
  __builtin_amdgcn_tensor_load_to_lds(g0, g1, z, z, z8, 0);
#else
  __builtin_amdgcn_tensor_load_to_lds(g0, g1, z, z, 0);
#endif
}

// ---------------------------------------------------------------------------
// fp32 rows -> interleaved bf16 hi|lo rows ([0,256)=hi, [256,512)=lo) and
// exact fp32 row sq-norms.  One wave per row.
// ---------------------------------------------------------------------------
__global__ void nn_prep_rows(const float* __restrict__ src,
                             uint16_t* __restrict__ dhl,
                             float* __restrict__ sq, int nrows) {
  const int wave = threadIdx.x >> 5;
  const int lane = threadIdx.x & 31;
  const int row  = blockIdx.x * 8 + wave;
  if (row >= nrows) return;
  const float* s = src + (size_t)row * K_DIM;
  uint16_t* ph = dhl + (size_t)row * ROWW;         // hi half
  uint16_t* pl = ph + K_DIM;                       // lo half
  float acc = 0.f;
#pragma unroll
  for (int i = 0; i < K_DIM / 32; ++i) {
    const int k = lane + i * 32;
    const float f = s[k];
    acc = fmaf(f, f, acc);
    const uint32_t u  = __float_as_uint(f);        // RNE fp32 -> bf16
    const uint32_t hb = (u + 0x7FFFu + ((u >> 16) & 1u)) >> 16;
    const float    fh = __uint_as_float(hb << 16);
    const float    lo = f - fh;
    const uint32_t ul = __float_as_uint(lo);
    const uint32_t lb = (ul + 0x7FFFu + ((ul >> 16) & 1u)) >> 16;
    ph[k] = (uint16_t)hb;
    pl[k] = (uint16_t)lb;
  }
#pragma unroll
  for (int off = 16; off > 0; off >>= 1)
    acc += __shfl_xor(acc, off, 32);
  if (lane == 0) sq[row] = acc;
}

// ---------------------------------------------------------------------------
// Fused GEMM + argmin.  grid = (M_TOT/BM, NSPLIT), block = 128 (4 waves).
// Wave-resident A fragments; TDM double-buffered code tiles in LDS;
// dot = Ah*Bh + Ah*Bl + Al*Bh via 3 independent WMMA chains.
// ---------------------------------------------------------------------------
__global__ __launch_bounds__(128)
void nn_main(const uint16_t* __restrict__ xhl, const uint16_t* __restrict__ chl,
             const float* __restrict__ c2, const unsigned char* __restrict__ act,
             float* __restrict__ pmin, int* __restrict__ pidx) {
  __shared__ uint16_t smem[2 * TILE_U16];   // two 16KB tile buffers

  const int lane = threadIdx.x & 31;
  const int wave = threadIdx.x >> 5;
  const int half = lane >> 4;
  const int l16  = lane & 15;
  const int m_base = blockIdx.x * BM + wave * 16;
  const int nslice = blockIdx.y;
  const int n0     = nslice * NPS;

  const uint32_t lds_base = (uint32_t)(uintptr_t)(void*)&smem[0]; // LDS byte addr
  const uint64_t gbase    = (uint64_t)(uintptr_t)(const void*)chl;

  // --- persistent A fragments (ISA 16-bit A 16x32 layout) ------------------
  v16bf Ah[8], Al[8];
  {
    const int m = m_base + l16;                       // lane = M row
    const uint4* bq = (const uint4*)(xhl + (size_t)m * ROWW);
#pragma unroll
    for (int c = 0; c < 8; ++c) {
      const int e0 = c * 32 + half * 8;               // first K element
      FragB fa, fb;
      fa.q[0] = bq[e0 >> 3];                          // hi: K = e0..e0+7
      fa.q[1] = bq[(e0 + 16) >> 3];                   //     K = e0+16..e0+23
      fb.q[0] = bq[(K_DIM + e0) >> 3];                // lo half of the row
      fb.q[1] = bq[(K_DIM + e0 + 16) >> 3];
      Ah[c] = fa.v;
      Al[c] = fb.v;
    }
  }

  float minv[8];
  int   mini[8];
#pragma unroll
  for (int j = 0; j < 8; ++j) { minv[j] = __builtin_inff(); mini[j] = -1; }

  // prologue: kick tile 0 into buffer 0
  if (wave == 0)
    tdm_load_tile(lds_base, gbase + (uint64_t)(n0) * (ROWW * 2));

  for (int t = 0; t < TILES; ++t) {
    const int cur = t & 1;
    const int nt  = n0 + t * TN;

    if (wave == 0) {
      if (t + 1 < TILES) {
        // issue next tile into the other buffer, then wait for current tile
        tdm_load_tile(lds_base + (uint32_t)((1 - cur) * TILE_BYTES),
                      gbase + (uint64_t)(nt + TN) * (ROWW * 2));
        __builtin_amdgcn_s_wait_tensorcnt(1);
      } else {
        __builtin_amdgcn_s_wait_tensorcnt(0);
      }
    }
    __syncthreads();   // buf[cur] ready for all waves

    const uint16_t* tile = smem + cur * TILE_U16;
    v8f hh = {}, hl = {}, lh = {};
#pragma unroll
    for (int c = 0; c < 8; ++c) {
      // ISA 16-bit B 32x16 layout: lane%16 = N (code), half picks K 8h..8h+7 and 16+8h..16+8h+7
      const int e0 = l16 * ROWW + c * 32 + half * 8;
      FragB bh, bl;
      bh.q[0] = *(const uint4*)(tile + e0);
      bh.q[1] = *(const uint4*)(tile + e0 + 16);
      bl.q[0] = *(const uint4*)(tile + K_DIM + e0);
      bl.q[1] = *(const uint4*)(tile + K_DIM + e0 + 16);
      hh = __builtin_amdgcn_wmma_f32_16x16x32_bf16(false, Ah[c], false, bh.v,
                                                   (short)0, hh, false, false);
      hl = __builtin_amdgcn_wmma_f32_16x16x32_bf16(false, Ah[c], false, bl.v,
                                                   (short)0, hl, false, false);
      lh = __builtin_amdgcn_wmma_f32_16x16x32_bf16(false, Al[c], false, bh.v,
                                                   (short)0, lh, false, false);
    }
    __syncthreads();   // all waves done reading buf[cur]; it may be refilled

    // C/D layout: lane%16 = column n, VGPR j = row (half*8 + j)
    const int   n   = nt + l16;
    const float c2n = c2[n];
    const bool  a   = act[n] != 0;
#pragma unroll
    for (int j = 0; j < 8; ++j) {
      const float dot = hh[j] + hl[j] + lh[j];
      const float s   = c2n - 2.0f * dot;     // x2 added in final pass
      if (a && s < minv[j]) { minv[j] = s; mini[j] = n; }
    }
  }

  // butterfly min+argmin across the 16 columns held by each half-wave
#pragma unroll
  for (int off = 1; off < 16; off <<= 1) {
#pragma unroll
    for (int j = 0; j < 8; ++j) {
      const float ov = __shfl_xor(minv[j], off, 32);
      const int   oi = __shfl_xor(mini[j], off, 32);
      const bool take = (ov < minv[j]) ||
                        (ov == minv[j] && (uint32_t)oi < (uint32_t)mini[j]);
      if (take) { minv[j] = ov; mini[j] = oi; }
    }
  }

  if (l16 == 0) {
#pragma unroll
    for (int j = 0; j < 8; ++j) {
      const int row = m_base + half * 8 + j;
      pmin[(size_t)row * NSPLIT + nslice] = minv[j];
      pidx[(size_t)row * NSPLIT + nslice] = mini[j];
    }
  }
}

// ---------------------------------------------------------------------------
// Combine the NSPLIT slice partials, add x2, apply threshold, emit outputs.
// ---------------------------------------------------------------------------
__global__ void nn_final(const float* __restrict__ pmin, const int* __restrict__ pidx,
                         const float* __restrict__ x2,
                         int* __restrict__ oidx, float* __restrict__ omin) {
  const int m = blockIdx.x * blockDim.x + threadIdx.x;
  if (m >= M_TOT) return;
  float best = __builtin_inff();
  int   bi   = -1;
#pragma unroll
  for (int s = 0; s < NSPLIT; ++s) {
    const float v = pmin[(size_t)m * NSPLIT + s];
    const int   i = pidx[(size_t)m * NSPLIT + s];
    if (v < best) { best = v; bi = i; }   // ascending n => first-wins on ties
  }
  const float mind = x2[m] + best;
  oidx[m] = (mind <= THRESH) ? bi : -1;
  omin[m] = mind;
}

// ---------------------------------------------------------------------------
extern "C" void kernel_launch(void* const* d_in, const int* in_sizes, int n_in,
                              void* d_out, int out_size, void* d_ws, size_t ws_size,
                              hipStream_t stream) {
  const float*         x     = (const float*)d_in[0];          // [4096, 256]
  const float*         codes = (const float*)d_in[1];          // [16384, 256]
  const unsigned char* act   = (const unsigned char*)d_in[2];  // [16384] bool

  char* ws = (char*)d_ws;
  size_t off = 0;
  auto alloc = [&](size_t bytes) -> char* {
    char* p = ws + off;
    off = (off + bytes + 255) & ~(size_t)255;
    return p;
  };
  uint16_t* xhl = (uint16_t*)alloc((size_t)M_TOT * ROWW * 2);
  uint16_t* chl = (uint16_t*)alloc((size_t)N_TOT * ROWW * 2);
  float* x2   = (float*)alloc((size_t)M_TOT * 4);
  float* c2   = (float*)alloc((size_t)N_TOT * 4);
  float* pmin = (float*)alloc((size_t)M_TOT * NSPLIT * 4);
  int*   pidx = (int*)alloc((size_t)M_TOT * NSPLIT * 4);
  (void)ws_size; (void)in_sizes; (void)n_in; (void)out_size;

  nn_prep_rows<<<M_TOT / 8, 256, 0, stream>>>(x, xhl, x2, M_TOT);
  nn_prep_rows<<<N_TOT / 8, 256, 0, stream>>>(codes, chl, c2, N_TOT);

  nn_main<<<dim3(M_TOT / BM, NSPLIT), 128, 0, stream>>>(xhl, chl, c2, act,
                                                        pmin, pidx);

  int* out_idx   = (int*)d_out;              // first 4096: int32 idxs
  float* out_min = (float*)d_out + M_TOT;    // next 4096: f32 min dists
  nn_final<<<M_TOT / 256, 256, 0, stream>>>(pmin, pidx, x2, out_idx, out_min);
}
